// SelfAttention_37220186587794
// MI455X (gfx1250) — matrix operand order repaired and checked by reading, not compiled, not run on hardware
//
#include <hip/hip_runtime.h>

// ---------------------------------------------------------------------------
// CDNA5 (gfx1250) self-attention, bf16 WMMA path, wave32, async-LDS staging.
//   SEQ=2048, B=2, H=4096, N_HEADS=32, N_KV=2, HEAD_DIM=128, ROT_DIM=64
// ---------------------------------------------------------------------------

typedef __attribute__((ext_vector_type(16))) __bf16 v16bf;
typedef __attribute__((ext_vector_type(8)))  float  v8f;

static constexpr int SEQ  = 2048;
static constexpr int BAT  = 2;
static constexpr int HDIM = 4096;
static constexpr int NH   = 32;
static constexpr int NKVH = 2;
static constexpr int DH   = 128;   // head dim
static constexpr int MROW = SEQ * BAT;          // 4096 rows (s*B + b)
static constexpr int KVW  = NKVH * DH;          // 256

__device__ __forceinline__ v8f zero8() {
  v8f z;
#pragma unroll
  for (int i = 0; i < 8; ++i) z[i] = 0.0f;
  return z;
}

__device__ __forceinline__ v8f wmma_bf16(v16bf a, v16bf b, v8f c) {
  // D(f32 16x16) = A(16x32 bf16) * B(32x16 bf16) + C
  return __builtin_amdgcn_wmma_f32_16x16x32_bf16(
      false, a, false, b, (short)0, c, false, false);
}

// Async global->LDS copy, 16 bytes per lane.  Tracked by ASYNCcnt.
// LDS byte offset = low 32 bits of the generic pointer to a __shared__ object.
__device__ __forceinline__ void async_ld_b128(void* lds, const void* gaddr) {
  unsigned int loff = (unsigned int)(unsigned long long)lds;
  asm volatile("global_load_async_to_lds_b128 %0, %1, off"
               :: "v"(loff), "v"(gaddr)
               : "memory");
}
__device__ __forceinline__ void wait_async0() {
  asm volatile("s_wait_asynccnt 0x0" ::: "memory");
}

// A-operand fragment (16x32 bf16): lane = M row; lanes0-15: K 0..15,
// lanes16-31: K 16..31.  Two 16-byte LDS reads.
__device__ __forceinline__ v16bf load_fragA(const __bf16* base, int lane, int stride) {
  const __bf16* p = base + (lane & 15) * stride + ((lane & 16) ? 16 : 0);
  union { uint4 u[2]; v16bf v; } t;
  t.u[0] = *(const uint4*)(p);
  t.u[1] = *(const uint4*)(p + 8);
  return t.v;
}

// B-operand fragment (32x16 bf16): lane = N col; lanes0-15: K 0..15,
// lanes16-31: K 16..31.  Two 16-byte LDS reads.
__device__ __forceinline__ v16bf load_fragB(const __bf16* base, int lane, int stride) {
  const __bf16* p = base + (lane & 15) * stride + ((lane & 16) ? 16 : 0);
  union { uint4 u[2]; v16bf v; } t;
  t.u[0] = *(const uint4*)(p);
  t.u[1] = *(const uint4*)(p + 8);
  return t.v;
}

// ---------------------------------------------------------------------------
// f32 -> bf16 elementwise convert
// ---------------------------------------------------------------------------
__global__ __launch_bounds__(256) void cvt_f32_bf16(
    const float* __restrict__ in, __bf16* __restrict__ out, size_t n) {
  size_t i = (size_t)blockIdx.x * 256 + threadIdx.x;
  if (i < n) out[i] = (__bf16)in[i];
}

// ---------------------------------------------------------------------------
// W[K][N] f32 -> Wt[N][K] bf16   (32x32 LDS tile transpose)
// ---------------------------------------------------------------------------
__global__ __launch_bounds__(256) void transpose_to_bf16(
    const float* __restrict__ in, __bf16* __restrict__ out, int K, int N) {
  __shared__ float tile[32][33];
  int nn = blockIdx.x * 32, kk = blockIdx.y * 32;
  int tx = threadIdx.x & 31, ty = threadIdx.x >> 5;  // 32 x 8
#pragma unroll
  for (int j = 0; j < 4; ++j) {
    int y = ty * 4 + j;
    tile[y][tx] = in[(size_t)(kk + y) * N + nn + tx];
  }
  __syncthreads();
#pragma unroll
  for (int j = 0; j < 4; ++j) {
    int y = ty * 4 + j;
    out[(size_t)(nn + y) * K + kk + tx] = (__bf16)tile[tx][y];
  }
}

// ---------------------------------------------------------------------------
// GEMM: C[M][ldc] = A[M][K](bf16) * Bt[N][K](bf16)^T + bias
// block = 256 thr (8 waves), tile 128x128, k-step 32, double-buffered LDS
// staged with global_load_async_to_lds_b128 (ASYNCcnt).
// wave(wm 0..3, wn 0..1) computes 32x64 = 2x4 WMMA frags
// ---------------------------------------------------------------------------
__global__ __launch_bounds__(256) void gemm_bf16_tn(
    const __bf16* __restrict__ A, const __bf16* __restrict__ Bt,
    const float* __restrict__ bias, float* __restrict__ C,
    int K, int ldc) {
  __shared__ __align__(16) __bf16 As[2][128 * 32];
  __shared__ __align__(16) __bf16 Bs[2][128 * 32];

  const int tid = threadIdx.x, lane = tid & 31, wid = tid >> 5;
  const int wm = wid >> 1, wn = wid & 1;
  const size_t tm = (size_t)blockIdx.y * 128;
  const size_t tn = (size_t)blockIdx.x * 128;

  v8f acc[2][4];
#pragma unroll
  for (int i = 0; i < 2; ++i)
#pragma unroll
    for (int j = 0; j < 4; ++j) acc[i][j] = zero8();

  const int row  = tid >> 1;            // 0..127
  const int eoff = (tid & 1) * 16;      // element offset within 32-elem chunk
  const __bf16* gA = A  + (tm + row) * (size_t)K + eoff;
  const __bf16* gB = Bt + (tn + row) * (size_t)K + eoff;
  __bf16* lA0 = &As[0][row * 32 + eoff];
  __bf16* lB0 = &Bs[0][row * 32 + eoff];
  __bf16* lA1 = &As[1][row * 32 + eoff];
  __bf16* lB1 = &Bs[1][row * 32 + eoff];

  auto issue = [&](int kk, int buf) {
    __bf16* la = buf ? lA1 : lA0;
    __bf16* lb = buf ? lB1 : lB0;
    async_ld_b128(la,     gA + kk);
    async_ld_b128(la + 8, gA + kk + 8);
    async_ld_b128(lb,     gB + kk);
    async_ld_b128(lb + 8, gB + kk + 8);
  };

  const int nk = K >> 5;
  issue(0, 0);
  for (int i = 0; i < nk; ++i) {
    wait_async0();       // stage i resident in LDS
    __syncthreads();     // all waves done reading buffer (i+1)&1 (stage i-1)
    if (i + 1 < nk) issue((i + 1) << 5, (i + 1) & 1);  // overlaps WMMAs below

    const __bf16* as = As[i & 1];
    const __bf16* bs = Bs[i & 1];
    v16bf af0 = load_fragA(as + (wm * 32) * 32,      lane, 32);
    v16bf af1 = load_fragA(as + (wm * 32 + 16) * 32, lane, 32);
#pragma unroll
    for (int fn = 0; fn < 4; ++fn) {
      v16bf bf = load_fragB(bs + (wn * 64 + fn * 16) * 32, lane, 32);
      acc[0][fn] = wmma_bf16(af0, bf, acc[0][fn]);
      acc[1][fn] = wmma_bf16(af1, bf, acc[1][fn]);
    }
  }

#pragma unroll
  for (int fm = 0; fm < 2; ++fm)
#pragma unroll
    for (int fn = 0; fn < 4; ++fn)
#pragma unroll
      for (int r = 0; r < 8; ++r) {
        size_t m = tm + wm * 32 + fm * 16 + r + ((lane & 16) ? 8 : 0);
        size_t n = tn + wn * 64 + fn * 16 + (lane & 15);
        float v = acc[fm][fn][r];
        if (bias) v += bias[n];
        C[m * (size_t)ldc + n] = v;
      }
}

// ---------------------------------------------------------------------------
// RoPE + scale + layout pack:
//   Qf[M][4096] -> Qb[b][h][s][128]     (scaled by 1/sqrt(128))
//   Kf[M][256]  -> Kb[b][kv][t][128]
//   Vf[M][256]  -> Vt[b][kv][d][2048]   (transposed for PV wmma)
// one block per row (s*B+b)
// ---------------------------------------------------------------------------
__global__ __launch_bounds__(256) void rope_pack(
    const float* __restrict__ Qf, const float* __restrict__ Kf,
    const float* __restrict__ Vf, const float* __restrict__ rope,
    __bf16* __restrict__ Qb, __bf16* __restrict__ Kb, __bf16* __restrict__ Vt) {
  const int rowi = blockIdx.x;       // 0..4095
  const int s = rowi >> 1, b = rowi & 1;
  const int tid = threadIdx.x;
  const float scale = 0.08838834764831845f;  // 1/sqrt(128)

  // Q: 2048 pairs
  for (int i = tid; i < HDIM / 2; i += 256) {
    const int d2 = (2 * i) & (DH - 1);
    const int head = (2 * i) >> 7;
    float x0 = Qf[(size_t)rowi * HDIM + 2 * i];
    float x1 = Qf[(size_t)rowi * HDIM + 2 * i + 1];
    float o0 = x0, o1 = x1;
    if (d2 < 64) {
      float r0 = rope[(size_t)s * 64 + d2];
      float r1 = rope[(size_t)s * 64 + d2 + 1];
      o0 = x0 * r0 - x1 * r1;
      o1 = x1 * r0 + x0 * r1;
    }
    size_t base = (((size_t)b * NH + head) * SEQ + s) * DH + d2;
    Qb[base]     = (__bf16)(o0 * scale);
    Qb[base + 1] = (__bf16)(o1 * scale);
  }
  // K: 128 pairs
  if (tid < KVW / 2) {
    const int i = tid;
    const int d2 = (2 * i) & (DH - 1);
    const int kv = (2 * i) >> 7;
    float x0 = Kf[(size_t)rowi * KVW + 2 * i];
    float x1 = Kf[(size_t)rowi * KVW + 2 * i + 1];
    float o0 = x0, o1 = x1;
    if (d2 < 64) {
      float r0 = rope[(size_t)s * 64 + d2];
      float r1 = rope[(size_t)s * 64 + d2 + 1];
      o0 = x0 * r0 - x1 * r1;
      o1 = x1 * r0 + x0 * r1;
    }
    size_t base = (((size_t)b * NKVH + kv) * SEQ + s) * DH + d2;
    Kb[base]     = (__bf16)o0;
    Kb[base + 1] = (__bf16)o1;
  }
  // V: 256 elems, transposed store
  {
    const int kv = tid >> 7, d = tid & (DH - 1);
    float v = Vf[(size_t)rowi * KVW + tid];
    Vt[(((size_t)b * NKVH + kv) * DH + d) * SEQ + s] = (__bf16)v;
  }
}

// ---------------------------------------------------------------------------
// Flash attention (causal, GQA), one (128-query tile, head, batch) per block.
// 8 waves; wave w owns 16 queries.  Key blocks of 32, async double-buffered.
//   S^T = K * Q^T  (softmax over keys is in-lane + one xor-16 shuffle)
//   ctx^T += V^T * P^T
// ---------------------------------------------------------------------------
__global__ __launch_bounds__(256) void flash_attn(
    const __bf16* __restrict__ Qb, const __bf16* __restrict__ Kb,
    const __bf16* __restrict__ Vt, __bf16* __restrict__ Ctx) {
  constexpr int BQ = 128, BT = 32;
  const int s0  = blockIdx.x * BQ;
  const int h   = blockIdx.y;
  const int b   = blockIdx.z;
  const int kv  = h >> 4;                 // 32 heads / 2 kv heads
  const int tid = threadIdx.x, lane = tid & 31, w = tid >> 5;

  __shared__ __align__(16) __bf16 Qs[BQ * DH];      // 32 KB
  __shared__ __align__(16) __bf16 Ks[2][BT * DH];   // 16 KB
  __shared__ __align__(16) __bf16 Vs[2][DH * BT];   // 16 KB

  const __bf16* ksrc = Kb + ((size_t)b * NKVH + kv) * SEQ * DH;
  const __bf16* vsrc = Vt + ((size_t)b * NKVH + kv) * DH * SEQ;

  // stage Q tile (coalesced: 2 threads per 256B row)
  {
    const __bf16* q = Qb + (((size_t)b * NH + h) * SEQ + s0) * DH;
    int row = tid >> 1, c = (tid & 1) * 8;
    const uint4* g = (const uint4*)(q + (size_t)row * DH);
    uint4* l = (uint4*)(Qs + row * DH);
#pragma unroll
    for (int i = 0; i < 8; ++i) l[c + i] = g[c + i];
  }

  // async prologue for key-block 0
  const int krow = tid >> 3, kcol = (tid & 7) * 16;   // K tile: 32 x 128
  const int vrow = tid >> 1, vcol = (tid & 1) * 16;   // V^T tile: 128 x 32
  auto issueKV = [&](int tb, int buf) {
    __bf16* lk = &Ks[buf][krow * DH + kcol];
    __bf16* lv = &Vs[buf][vrow * BT + vcol];
    async_ld_b128(lk,     ksrc + (size_t)(tb + krow) * DH + kcol);
    async_ld_b128(lk + 8, ksrc + (size_t)(tb + krow) * DH + kcol + 8);
    async_ld_b128(lv,     vsrc + (size_t)vrow * SEQ + tb + vcol);
    async_ld_b128(lv + 8, vsrc + (size_t)vrow * SEQ + tb + vcol + 8);
  };
  issueKV(0, 0);

  __syncthreads();   // Q tile visible

  // hoist the 4 loop-invariant Q B-operand fragments into registers
  v16bf bQ[4];
#pragma unroll
  for (int kd = 0; kd < 4; ++kd)
    bQ[kd] = load_fragB(Qs + (w * 16) * DH + kd * 32, lane, DH);

  v8f acc[8];
#pragma unroll
  for (int i = 0; i < 8; ++i) acc[i] = zero8();
  float mrun = -1e30f, lrun = 0.0f;

  const int slocal = w * 16 + (lane & 15);
  const int sidx   = s0 + slocal;
  const int thalf  = (lane & 16) ? 8 : 0;

  const int nb = (s0 + BQ) / BT;
  for (int ib = 0; ib < nb; ++ib) {
    const int tb = ib * BT;
    wait_async0();      // key-block ib resident in LDS
    __syncthreads();    // all waves done with buffer (ib+1)&1
    if (ib + 1 < nb) issueKV(tb + BT, (ib + 1) & 1);  // overlaps math below

    const __bf16* ks = Ks[ib & 1];
    const __bf16* vs = Vs[ib & 1];

    // S^T fragments: M = t (two 16-row frags), N = s (this wave's 16 queries)
    v8f st0 = zero8(), st1 = zero8();
#pragma unroll
    for (int kd = 0; kd < 4; ++kd) {
      v16bf aK0 = load_fragA(ks + kd * 32,           lane, DH);
      v16bf aK1 = load_fragA(ks + 16 * DH + kd * 32, lane, DH);
      st0 = wmma_bf16(aK0, bQ[kd], st0);
      st1 = wmma_bf16(aK1, bQ[kd], st1);
    }

    // causal mask + online softmax (per lane: 16 scores for this s)
    float p0[8], p1[8], bmax = -1e30f;
#pragma unroll
    for (int r = 0; r < 8; ++r) {
      int t0 = tb + thalf + r;
      float v0 = (t0 > sidx) ? -1e30f : st0[r];
      int t1 = tb + 16 + thalf + r;
      float v1 = (t1 > sidx) ? -1e30f : st1[r];
      p0[r] = v0; p1[r] = v1;
      bmax = fmaxf(bmax, fmaxf(v0, v1));
    }
    bmax = fmaxf(bmax, __shfl_xor(bmax, 16));
    float mnew = fmaxf(mrun, bmax);
    float corr = __expf(mrun - mnew);
    mrun = mnew;
    float ps = 0.0f;
#pragma unroll
    for (int r = 0; r < 8; ++r) {
      p0[r] = __expf(p0[r] - mnew);
      p1[r] = __expf(p1[r] - mnew);
      ps += p0[r] + p1[r];
    }
    ps += __shfl_xor(ps, 16);
    lrun = lrun * corr + ps;
#pragma unroll
    for (int i = 0; i < 8; ++i)
#pragma unroll
      for (int j = 0; j < 8; ++j) acc[i][j] *= corr;

    // Convert P^T D-frags -> B-operand fragment for V^T * P^T (xor-16 shuffles)
    v16bf pB;
#pragma unroll
    for (int r = 0; r < 8; ++r) {
      float x0 = __shfl_xor(p0[r], 16);
      float x1 = __shfl_xor(p1[r], 16);
      float lo = (lane & 16) ? x1 : p0[r];   // K(t) = tb + e | tb + 16 + e
      float hi = (lane & 16) ? p1[r] : x0;
      pB[r]     = (__bf16)lo;
      pB[r + 8] = (__bf16)hi;
    }

    // ctx^T[d][s] accumulate
#pragma unroll
    for (int df = 0; df < 8; ++df) {
      v16bf aV = load_fragA(vs + df * 16 * BT, lane, BT);
      acc[df] = wmma_bf16(aV, pB, acc[df]);
    }
  }

  // normalize + transpose back through LDS, write bf16 ctx[M][4096]
  const float inv = 1.0f / lrun;
  __syncthreads();
#pragma unroll
  for (int df = 0; df < 8; ++df)
#pragma unroll
    for (int r = 0; r < 8; ++r) {
      int d = df * 16 + r + ((lane & 16) ? 8 : 0);
      Qs[slocal * DH + d] = (__bf16)(acc[df][r] * inv);
    }
  __syncthreads();
  {
    int row = tid >> 1, c = (tid & 1) * 8;
    size_t grow = ((size_t)(s0 + row) * BAT + b) * HDIM + (size_t)h * DH;
    uint4* g = (uint4*)(Ctx + grow);
    const uint4* l = (const uint4*)(Qs + row * DH);
#pragma unroll
    for (int i = 0; i < 8; ++i) g[c + i] = l[c + i];
  }
}

// ---------------------------------------------------------------------------
extern "C" void kernel_launch(void* const* d_in, const int* in_sizes, int n_in,
                              void* d_out, int out_size, void* d_ws, size_t ws_size,
                              hipStream_t stream) {
  const float* hs   = (const float*)d_in[0];
  const float* rope = (const float*)d_in[1];
  const float* Wq   = (const float*)d_in[2];
  const float* bq   = (const float*)d_in[3];
  const float* Wk   = (const float*)d_in[4];
  const float* bk   = (const float*)d_in[5];
  const float* Wv   = (const float*)d_in[6];
  const float* bv   = (const float*)d_in[7];
  const float* Wo   = (const float*)d_in[8];
  float* out = (float*)d_out;

  char* cur = (char*)d_ws;
  auto alloc = [&](size_t bytes) -> char* {
    char* p = cur;
    cur += (bytes + 255) & ~(size_t)255;
    return p;
  };
  __bf16* Xbf = (__bf16*)alloc((size_t)MROW * HDIM * 2);
  __bf16* WqT = (__bf16*)alloc((size_t)HDIM * HDIM * 2);
  __bf16* WkT = (__bf16*)alloc((size_t)KVW * HDIM * 2);
  __bf16* WvT = (__bf16*)alloc((size_t)KVW * HDIM * 2);
  __bf16* WoT = (__bf16*)alloc((size_t)HDIM * HDIM * 2);
  float*  Qf  = (float*)alloc((size_t)MROW * HDIM * 4);
  float*  Kf  = (float*)alloc((size_t)MROW * KVW * 4);
  float*  Vf  = (float*)alloc((size_t)MROW * KVW * 4);
  __bf16* Qb  = (__bf16*)alloc((size_t)MROW * HDIM * 2);
  __bf16* Kb  = (__bf16*)alloc((size_t)MROW * KVW * 2);
  __bf16* Vt  = (__bf16*)alloc((size_t)MROW * KVW * 2);
  __bf16* Cb  = (__bf16*)alloc((size_t)MROW * HDIM * 2);

  // 1) X -> bf16
  cvt_f32_bf16<<<(MROW * (size_t)HDIM) / 256, 256, 0, stream>>>(
      hs, Xbf, (size_t)MROW * HDIM);

  // 2) weight transposes -> bf16 [N][K]
  transpose_to_bf16<<<dim3(HDIM / 32, HDIM / 32), 256, 0, stream>>>(Wq, WqT, HDIM, HDIM);
  transpose_to_bf16<<<dim3(KVW  / 32, HDIM / 32), 256, 0, stream>>>(Wk, WkT, HDIM, KVW);
  transpose_to_bf16<<<dim3(KVW  / 32, HDIM / 32), 256, 0, stream>>>(Wv, WvT, HDIM, KVW);
  transpose_to_bf16<<<dim3(HDIM / 32, HDIM / 32), 256, 0, stream>>>(Wo, WoT, HDIM, HDIM);

  // 3) QKV projections (WMMA, async-staged)
  gemm_bf16_tn<<<dim3(HDIM / 128, MROW / 128), 256, 0, stream>>>(Xbf, WqT, bq, Qf, HDIM, HDIM);
  gemm_bf16_tn<<<dim3(KVW  / 128, MROW / 128), 256, 0, stream>>>(Xbf, WkT, bk, Kf, HDIM, KVW);
  gemm_bf16_tn<<<dim3(KVW  / 128, MROW / 128), 256, 0, stream>>>(Xbf, WvT, bv, Vf, HDIM, KVW);

  // 4) RoPE + scale + attention layouts
  rope_pack<<<MROW, 256, 0, stream>>>(Qf, Kf, Vf, rope, Qb, Kb, Vt);

  // 5) causal flash attention (WMMA, async-staged)
  flash_attn<<<dim3(SEQ / 128, NH, BAT), 256, 0, stream>>>(Qb, Kb, Vt, Cb);

  // 6) output projection (WMMA) -> d_out f32
  gemm_bf16_tn<<<dim3(HDIM / 128, MROW / 128), 256, 0, stream>>>(
      Cb, WoT, nullptr, out, HDIM, HDIM);
}
